// SparseGraphTransformerLayer_88527865905550
// MI455X (gfx1250) — matrix operand order repaired and checked by reading, not compiled, not run on hardware
//
#include <hip/hip_runtime.h>
#include <cstddef>
#include <cstdint>

// ---------------------------------------------------------------------------
// Types
// ---------------------------------------------------------------------------
typedef __attribute__((ext_vector_type(16))) __bf16 v16bf;
typedef __attribute__((ext_vector_type(8)))  __bf16 v8bf;
typedef __attribute__((ext_vector_type(8)))  float  v8f;
typedef __attribute__((ext_vector_type(4)))  unsigned int v4u;
typedef __attribute__((ext_vector_type(8)))  int    v8i;
typedef __attribute__((ext_vector_type(4)))  int    v4i;

#define B_  16
#define N_  512
#define D_  512
#define H_  8
#define HD_ 64
#define TOPK_ 16
#define ROWS_ (B_ * N_)       // 8192
#define FF_  (4 * D_)         // 2048

// GEMM macro-tile: 128 (M) x 64 (N), K-step 32, 8 waves (wave = M sub-tile)
#define GM_ 128
#define GN_ 64
#define LDS_A_BYTES (GM_ * 32 * 2)   // 8192
#define LDS_B_BYTES (GN_ * 32 * 2)   // 4096
#define LDS_BUF_BYTES (LDS_A_BYTES + LDS_B_BYTES)  // 12288
#define LDS_TOTAL (2 * LDS_BUF_BYTES)              // 24576

// ---------------------------------------------------------------------------
// Helpers
// ---------------------------------------------------------------------------
__device__ __forceinline__ float gelu_exact(float x) {
    return 0.5f * x * (1.0f + erff(x * 0.70710678118654752f));
}

// 16x32 bf16 A-fragment from a row-major buffer (leading dim ld, halves).
// lanes 0-15: M=lane, K=j (j=0..15) ; lanes 16-31: M=lane-16, K=16+j
__device__ __forceinline__ v16bf load_a16x32(const __bf16* tile, int ld, int lane) {
    int m  = lane & 15;
    int kb = (lane & 16) ? 16 : 0;
    const __bf16* p = tile + (size_t)m * ld + kb;
    v8bf lo = *(const v8bf*)(p);
    v8bf hi = *(const v8bf*)(p + 8);
    v16bf r;
#pragma unroll
    for (int j = 0; j < 8; ++j) { r[j] = lo[j]; r[j + 8] = hi[j]; }
    return r;
}

// ---------------------------------------------------------------------------
// TDM: tensor_load_to_lds (6-arg clang-23 builtin on this toolchain),
// D# per ISA ch.8. 2D tile: tile_d0 elements per row, tile_d1 rows, row
// stride stride0 (elems), bf16 data (data_size=1 -> 2 bytes). lds_off is a
// byte offset from LDS base 0 (kernel uses dynamic LDS only, so its
// allocation starts at offset 0).
// ---------------------------------------------------------------------------
#if __has_builtin(__builtin_amdgcn_tensor_load_to_lds)
#define HAVE_TDM 1
__device__ __forceinline__ void tdm_load_2d(unsigned lds_off, const void* gptr,
                                            unsigned tile_d0, unsigned tile_d1,
                                            unsigned tensor_d0, unsigned tensor_d1,
                                            unsigned long long stride0) {
    unsigned long long ga = (unsigned long long)gptr;
    v4u g0;
    g0[0] = 1u;                                   // count=1, user desc, no gather
    g0[1] = lds_off;                              // lds_addr (bytes)
    g0[2] = (unsigned)(ga & 0xFFFFFFFFu);         // global_addr[31:0]
    g0[3] = (unsigned)((ga >> 32) & 0x01FFFFFFu)  // global_addr[56:32]
          | (2u << 30);                           // type = 2 ("image")
    v8i g1;
    g1[0] = (int)(1u << 16);                      // wg_mask=0, data_size=1 (2B)
    g1[1] = (int)((tensor_d0 & 0xFFFFu) << 16);   // [63:48] = tensor_dim0[15:0]
    g1[2] = (int)(((tensor_d0 >> 16) & 0xFFFFu) | ((tensor_d1 & 0xFFFFu) << 16));
    g1[3] = (int)(((tensor_d1 >> 16) & 0xFFFFu) | ((tile_d0 & 0xFFFFu) << 16));
    g1[4] = (int)(tile_d1 & 0xFFFFu);             // tile_dim1, tile_dim2=0
    g1[5] = (int)(stride0 & 0xFFFFFFFFull);       // tensor_dim0_stride[31:0]
    g1[6] = (int)((stride0 >> 32) & 0xFFFFull);   // stride0[47:32], stride1=0
    g1[7] = 0;
    v4i z4 = {0, 0, 0, 0};                        // groups 2/3 unused (<=2D)
    v8i z8 = {0, 0, 0, 0, 0, 0, 0, 0};
    __builtin_amdgcn_tensor_load_to_lds(g0, g1, z4, z4, z8, 0);
}
__device__ __forceinline__ void tdm_wait0() {
#if __has_builtin(__builtin_amdgcn_s_wait_tensorcnt)
    __builtin_amdgcn_s_wait_tensorcnt((short)0);
#endif
}
#else
#define HAVE_TDM 0
#endif

// ---------------------------------------------------------------------------
// Weight pack: row-major fp32 W[Kd][N] -> bf16 WMMA B-fragment tiles, ordered
// [kt][tn] so one K-step's GN_/16 tiles are contiguous (single TDM transfer).
// Within a 32x16 tile, lane's 16 halves are contiguous: lanes 0-15 K=kt*32+j,
// lanes 16-31 K=kt*32+16+j, column n = tn*16 + (lane&15).
// ---------------------------------------------------------------------------
__global__ void pack_wb_kernel(const float* __restrict__ W, __bf16* __restrict__ out,
                               int Kd, int N) {
    int t = blockIdx.x * blockDim.x + threadIdx.x;
    int TN = N >> 4;
    int total = (Kd >> 5) * TN * 32;
    if (t >= total) return;
    int lane = t & 31;
    int tile = t >> 5;
    int tn = tile % TN;
    int kt = tile / TN;
    int n  = tn * 16 + (lane & 15);
    int k0 = kt * 32 + ((lane & 16) ? 16 : 0);
    __bf16* o = out + (size_t)tile * 512 + lane * 16;
#pragma unroll
    for (int j = 0; j < 16; ++j)
        o[j] = (__bf16)W[(size_t)(k0 + j) * N + n];
}

// ---------------------------------------------------------------------------
// Row LayerNorm (D=512) -> bf16, 256 threads/row
// ---------------------------------------------------------------------------
__global__ __launch_bounds__(256) void ln_bf16_kernel(const float* __restrict__ x,
                                                      const float* __restrict__ g,
                                                      const float* __restrict__ bt,
                                                      __bf16* __restrict__ out) {
    int row = blockIdx.x;
    int tid = threadIdx.x;
    const float* xr = x + (size_t)row * D_;
    float v0 = xr[tid], v1 = xr[tid + 256];
    __shared__ float rs[256], rss[256];
    rs[tid]  = v0 + v1;
    rss[tid] = v0 * v0 + v1 * v1;
    __syncthreads();
    for (int o = 128; o; o >>= 1) {
        if (tid < o) { rs[tid] += rs[tid + o]; rss[tid] += rss[tid + o]; }
        __syncthreads();
    }
    float mean = rs[0] * (1.0f / D_);
    float var  = rss[0] * (1.0f / D_) - mean * mean;
    float inv  = rsqrtf(var + 1e-5f);
    __bf16* orow = out + (size_t)row * D_;
    orow[tid]       = (__bf16)((v0 - mean) * inv * g[tid]       + bt[tid]);
    orow[tid + 256] = (__bf16)((v1 - mean) * inv * g[tid + 256] + bt[tid + 256]);
}

// ---------------------------------------------------------------------------
// WMMA GEMM, TDM-staged through LDS, double buffered.
// Block: 256 threads = 8 waves; macro tile 128x64. Wave w owns rows
// [w*16, w*16+16), 4 accumulators across the 64 output columns, so each
// A-fragment feeds 4 WMMAs. A panel (128x32) and B panel (32x64, pre-packed
// fragment order) are DMA'd into LDS by TDM from wave 0; waves consume via
// ds_load_b128 after s_wait_tensorcnt + barrier.
// ---------------------------------------------------------------------------
__global__ __launch_bounds__(256) void gemm_bf16_wmma(const __bf16* __restrict__ A,
                                                      const __bf16* __restrict__ Bp,
                                                      const float* __restrict__ bias,
                                                      const float* __restrict__ resid,
                                                      float* __restrict__ Cf,
                                                      __bf16* __restrict__ Cbf,
                                                      int M, int N, int Kd, int act) {
    extern __shared__ char smem_raw[];
    __bf16* smem = (__bf16*)smem_raw;

    int lane = threadIdx.x & 31;
    int wave = threadIdx.x >> 5;
    int tnc  = N / GN_;                 // macro tiles along N
    int tmm  = blockIdx.x / tnc;        // macro tile row
    int tnm  = blockIdx.x % tnc;        // macro tile col
    int KT   = Kd >> 5;
    int TN   = N >> 4;

    const __bf16* Abase = A + (size_t)tmm * GM_ * Kd;            // 128 rows
    const __bf16* Bbase = Bp + ((size_t)tnm * (GN_ / 16)) * 512; // +kt*TN*512

#if HAVE_TDM
    if (wave == 0) {
        tdm_load_2d(0,            Abase, 32, GM_, (unsigned)Kd, GM_, (unsigned long long)Kd);
        tdm_load_2d(LDS_A_BYTES,  Bbase, 2048, 1, 2048, 1, 2048ull);
        tdm_wait0();
    }
    __syncthreads();
#else
    { // cooperative fallback copy of panel kt=0
        for (int i = threadIdx.x; i < GM_ * 32 / 8; i += 256) {
            int r = i / 4, cchunk = (i & 3) * 8;
            *(v8bf*)(smem + r * 32 + cchunk) = *(const v8bf*)(Abase + (size_t)r * Kd + cchunk);
        }
        for (int i = threadIdx.x; i < 2048 / 8; i += 256)
            *(v8bf*)(smem + LDS_A_BYTES / 2 + i * 8) = *(const v8bf*)(Bbase + i * 8);
        __syncthreads();
    }
#endif

    v8f c0 = {}, c1 = {}, c2 = {}, c3 = {};
    int m  = lane & 15;
    int kb = (lane & 16) ? 16 : 0;

    for (int kt = 0; kt < KT; ++kt) {
        int cur = kt & 1;
        const __bf16* As = smem + (size_t)cur * (LDS_BUF_BYTES / 2);
        const __bf16* Bs = As + LDS_A_BYTES / 2;

#if HAVE_TDM
        if (wave == 0 && kt + 1 < KT) {
            int nxt = (kt + 1) & 1;
            tdm_load_2d((unsigned)(nxt * LDS_BUF_BYTES),
                        Abase + (kt + 1) * 32, 32, GM_, (unsigned)Kd, GM_,
                        (unsigned long long)Kd);
            tdm_load_2d((unsigned)(nxt * LDS_BUF_BYTES + LDS_A_BYTES),
                        Bbase + (size_t)(kt + 1) * TN * 512, 2048, 1, 2048, 1, 2048ull);
        }
#endif
        // A fragment for this wave's 16 rows (LDS, ld=32)
        const __bf16* ap = As + ((size_t)(wave * 16 + m)) * 32 + kb;
        v8bf lo = *(const v8bf*)(ap);
        v8bf hi = *(const v8bf*)(ap + 8);
        v16bf a;
#pragma unroll
        for (int j = 0; j < 8; ++j) { a[j] = lo[j]; a[j + 8] = hi[j]; }

        v16bf b0 = *(const v16bf*)(Bs + 0 * 512 + lane * 16);
        v16bf b1 = *(const v16bf*)(Bs + 1 * 512 + lane * 16);
        v16bf b2 = *(const v16bf*)(Bs + 2 * 512 + lane * 16);
        v16bf b3 = *(const v16bf*)(Bs + 3 * 512 + lane * 16);
        c0 = __builtin_amdgcn_wmma_f32_16x16x32_bf16(false, a, false, b0, (short)0, c0, false, false);
        c1 = __builtin_amdgcn_wmma_f32_16x16x32_bf16(false, a, false, b1, (short)0, c1, false, false);
        c2 = __builtin_amdgcn_wmma_f32_16x16x32_bf16(false, a, false, b2, (short)0, c2, false, false);
        c3 = __builtin_amdgcn_wmma_f32_16x16x32_bf16(false, a, false, b3, (short)0, c3, false, false);

#if HAVE_TDM
        if (wave == 0 && kt + 1 < KT) tdm_wait0();
        __syncthreads();
#else
        __syncthreads();
        if (kt + 1 < KT) {
            const __bf16* An = Abase + (kt + 1) * 32;
            const __bf16* Bn = Bbase + (size_t)(kt + 1) * TN * 512;
            for (int i = threadIdx.x; i < GM_ * 32 / 8; i += 256) {
                int r = i / 4, cchunk = (i & 3) * 8;
                *(v8bf*)(smem + r * 32 + cchunk) = *(const v8bf*)(An + (size_t)r * Kd + cchunk);
            }
            for (int i = threadIdx.x; i < 2048 / 8; i += 256)
                *(v8bf*)(smem + LDS_A_BYTES / 2 + i * 8) = *(const v8bf*)(Bn + i * 8);
            __syncthreads();
        }
#endif
    }

    // Epilogue: 16x64 per wave
    int rbase = (lane & 16) ? 8 : 0;
    int nl    = lane & 15;
    v8f acc[4] = {c0, c1, c2, c3};
#pragma unroll
    for (int nf = 0; nf < 4; ++nf) {
        int col  = tnm * GN_ + nf * 16 + nl;
        float bv = bias ? bias[col] : 0.0f;
#pragma unroll
        for (int r = 0; r < 8; ++r) {
            int row = tmm * GM_ + wave * 16 + rbase + r;
            float v = acc[nf][r] + bv;
            if (act) v = gelu_exact(v);
            if (resid) v += resid[(size_t)row * N + col];
            if (Cf)  Cf[(size_t)row * N + col]  = v;
            if (Cbf) Cbf[(size_t)row * N + col] = (__bf16)v;
        }
    }
}

// ---------------------------------------------------------------------------
// Fused sparse attention: one block per (b, h, 16-query tile).
// WMMA QK^T tile -> LDS logits; per-row wave-wide top-16 extraction; softmax;
// sparse gather of V. Output bf16 row-major (b,n,h,hd).
// ---------------------------------------------------------------------------
__global__ __launch_bounds__(128) void attn_topk_fused(const __bf16* __restrict__ Qb,
                                                       const __bf16* __restrict__ Kb,
                                                       const __bf16* __restrict__ Vb,
                                                       const float* __restrict__ rel,
                                                       __bf16* __restrict__ attnb) {
    __shared__ float lg[16][N_];   // 32 KB logits tile
    int lane = threadIdx.x & 31;
    int wave = threadIdx.x >> 5;
    int qt = blockIdx.x & 31;
    int h  = (blockIdx.x >> 5) & 7;
    int b  = blockIdx.x >> 8;

    const size_t qoff = ((size_t)b * N_ + qt * 16) * D_ + h * HD_;
    v16bf a0 = load_a16x32(Qb + qoff,      D_, lane);
    v16bf a1 = load_a16x32(Qb + qoff + 32, D_, lane);

    int nl   = lane & 15;
    int kb16 = (lane & 16) ? 16 : 0;
    int rb   = (lane & 16) ? 8 : 0;

    for (int nt = wave; nt < (N_ / 16); nt += 4) {
        const __bf16* kp = Kb + ((size_t)b * N_ + nt * 16 + nl) * D_ + h * HD_ + kb16;
        v16bf b0 = *(const v16bf*)kp;
        v16bf b1 = *(const v16bf*)(kp + 32);
        v8f c = {};
        c = __builtin_amdgcn_wmma_f32_16x16x32_bf16(false, a0, false, b0, (short)0, c, false, false);
        c = __builtin_amdgcn_wmma_f32_16x16x32_bf16(false, a1, false, b1, (short)0, c, false, false);
        int nk = nt * 16 + nl;
#pragma unroll
        for (int r = 0; r < 8; ++r) {
            int m  = rb + r;
            int nq = qt * 16 + m;
            lg[m][nk] = c[r] * 0.125f + rel[(size_t)(nk - nq + (N_ - 1)) * H_ + h];
        }
    }
    __syncthreads();

    for (int m = wave; m < 16; m += 4) {
        int nq = qt * 16 + m;
        float myv = 0.0f; int myi = 0;
#pragma unroll 1
        for (int it = 0; it < TOPK_; ++it) {
            float bv = -3.0e38f; int bi = 0;
#pragma unroll
            for (int t = 0; t < 16; ++t) {
                int j = lane + t * 32;
                float v = lg[m][j];
                if (v > bv) { bv = v; bi = j; }
            }
#pragma unroll
            for (int o = 16; o; o >>= 1) {
                float ov = __shfl_xor(bv, o);
                int   oi = __shfl_xor(bi, o);
                if (ov > bv || (ov == bv && oi < bi)) { bv = ov; bi = oi; }
            }
            if (lane == it) { myv = bv; myi = bi; }
            if (lane == 0)  lg[m][bi] = -3.0e38f;
        }
        float m0 = __shfl(myv, 0);
        float pv = (lane < TOPK_) ? __expf(myv - m0) : 0.0f;
        float s = pv;
#pragma unroll
        for (int o = 16; o; o >>= 1) s += __shfl_xor(s, o);
        pv *= (1.0f / s);

        int hd = lane * 2;
        float acc0 = 0.0f, acc1 = 0.0f;
#pragma unroll
        for (int i = 0; i < TOPK_; ++i) {
            float pi = __shfl(pv, i);
            int   ni = __shfl(myi, i);
            const __bf16* vp = Vb + ((size_t)b * N_ + ni) * D_ + h * HD_ + hd;
            acc0 += pi * (float)vp[0];
            acc1 += pi * (float)vp[1];
        }
        __bf16* op = attnb + ((size_t)b * N_ + nq) * D_ + h * HD_ + hd;
        op[0] = (__bf16)acc0;
        op[1] = (__bf16)acc1;
    }
}

// ---------------------------------------------------------------------------
// Launch
// ---------------------------------------------------------------------------
extern "C" void kernel_launch(void* const* d_in, const int* in_sizes, int n_in,
                              void* d_out, int out_size, void* d_ws, size_t ws_size,
                              hipStream_t stream) {
    (void)in_sizes; (void)n_in; (void)out_size; (void)ws_size;

    const float* x   = (const float*)d_in[0];
    const float* wq  = (const float*)d_in[1];
    const float* bq  = (const float*)d_in[2];
    const float* wk  = (const float*)d_in[3];
    const float* bk  = (const float*)d_in[4];
    const float* wv  = (const float*)d_in[5];
    const float* bv  = (const float*)d_in[6];
    const float* wo  = (const float*)d_in[7];
    const float* bo  = (const float*)d_in[8];
    const float* g1  = (const float*)d_in[9];
    const float* be1 = (const float*)d_in[10];
    const float* g2  = (const float*)d_in[11];
    const float* be2 = (const float*)d_in[12];
    const float* w1  = (const float*)d_in[13];
    const float* bf1 = (const float*)d_in[14];
    const float* w2  = (const float*)d_in[15];
    const float* bf2 = (const float*)d_in[16];
    const float* rel = (const float*)d_in[17];

    float* out_f = (float*)d_out;
    char*  ws    = (char*)d_ws;

    const size_t SZ_DD  = (size_t)D_ * D_ * 2;
    const size_t SZ_DF  = (size_t)D_ * FF_ * 2;
    const size_t SZ_ACT = (size_t)ROWS_ * D_ * 2;
    size_t o = 0;
    size_t o_wq = o; o += SZ_DD;
    size_t o_wk = o; o += SZ_DD;
    size_t o_wv = o; o += SZ_DD;
    size_t o_wo = o; o += SZ_DD;
    size_t o_w1 = o; o += SZ_DF;
    size_t o_w2 = o; o += SZ_DF;
    size_t o_xn = o; o += SZ_ACT;   // xn/Q/K/V region (32 MB) reused as FFN h
    size_t o_q  = o; o += SZ_ACT;
    size_t o_k  = o; o += SZ_ACT;
    size_t o_v  = o; o += SZ_ACT;
    size_t o_at = o; o += SZ_ACT;   // attn out; reused as xn2
    size_t o_h  = o_xn;
    size_t o_x2 = o_at;

    __bf16* wq_p = (__bf16*)(ws + o_wq);
    __bf16* wk_p = (__bf16*)(ws + o_wk);
    __bf16* wv_p = (__bf16*)(ws + o_wv);
    __bf16* wo_p = (__bf16*)(ws + o_wo);
    __bf16* w1_p = (__bf16*)(ws + o_w1);
    __bf16* w2_p = (__bf16*)(ws + o_w2);
    __bf16* xn   = (__bf16*)(ws + o_xn);
    __bf16* Qb   = (__bf16*)(ws + o_q);
    __bf16* Kb   = (__bf16*)(ws + o_k);
    __bf16* Vb   = (__bf16*)(ws + o_v);
    __bf16* atb  = (__bf16*)(ws + o_at);
    __bf16* hb   = (__bf16*)(ws + o_h);
    __bf16* xn2  = (__bf16*)(ws + o_x2);

    // 1) pack weights into [kt][tn] WMMA B-fragment layout
    {
        int tdd = (D_ / 32) * (D_ / 16) * 32;
        int tdf = (D_ / 32) * (FF_ / 16) * 32;
        int tfd = (FF_ / 32) * (D_ / 16) * 32;
        pack_wb_kernel<<<(tdd + 255) / 256, 256, 0, stream>>>(wq, wq_p, D_, D_);
        pack_wb_kernel<<<(tdd + 255) / 256, 256, 0, stream>>>(wk, wk_p, D_, D_);
        pack_wb_kernel<<<(tdd + 255) / 256, 256, 0, stream>>>(wv, wv_p, D_, D_);
        pack_wb_kernel<<<(tdd + 255) / 256, 256, 0, stream>>>(wo, wo_p, D_, D_);
        pack_wb_kernel<<<(tdf + 255) / 256, 256, 0, stream>>>(w1, w1_p, D_, FF_);
        pack_wb_kernel<<<(tfd + 255) / 256, 256, 0, stream>>>(w2, w2_p, FF_, D_);
    }

    // 2) LN1
    ln_bf16_kernel<<<ROWS_, 256, 0, stream>>>(x, g1, be1, xn);

    // 3) Q, K, V projections: macro grid (8192/128)*(512/64) = 512 blocks
    int gDD = (ROWS_ / GM_) * (D_ / GN_);
    gemm_bf16_wmma<<<gDD, 256, LDS_TOTAL, stream>>>(xn, wq_p, bq, nullptr, nullptr, Qb,
                                                    ROWS_, D_, D_, 0);
    gemm_bf16_wmma<<<gDD, 256, LDS_TOTAL, stream>>>(xn, wk_p, bk, nullptr, nullptr, Kb,
                                                    ROWS_, D_, D_, 0);
    gemm_bf16_wmma<<<gDD, 256, LDS_TOTAL, stream>>>(xn, wv_p, bv, nullptr, nullptr, Vb,
                                                    ROWS_, D_, D_, 0);

    // 4) fused top-k attention: B*H*(N/16) = 4096 blocks
    attn_topk_fused<<<B_ * H_ * (N_ / 16), 128, 0, stream>>>(Qb, Kb, Vb, rel, atb);

    // 5) output projection + residual(x) -> d_out (f32)
    gemm_bf16_wmma<<<gDD, 256, LDS_TOTAL, stream>>>(atb, wo_p, bo, x, out_f, nullptr,
                                                    ROWS_, D_, D_, 0);

    // 6) LN2 on d_out -> xn2 (bf16)
    ln_bf16_kernel<<<ROWS_, 256, 0, stream>>>(out_f, g2, be2, xn2);

    // 7) FFN up + exact GELU: 8192 x 2048 x 512 -> 2048 blocks
    int gDF = (ROWS_ / GM_) * (FF_ / GN_);
    gemm_bf16_wmma<<<gDF, 256, LDS_TOTAL, stream>>>(xn2, w1_p, bf1, nullptr, nullptr, hb,
                                                    ROWS_, FF_, D_, 1);

    // 8) FFN down + residual(d_out) -> d_out: 8192 x 512 x 2048
    gemm_bf16_wmma<<<gDD, 256, LDS_TOTAL, stream>>>(hb, w2_p, bf2, out_f, out_f, nullptr,
                                                    ROWS_, D_, FF_, 0);
}
